// SONL_29429115912495
// MI455X (gfx1250) — matrix operand order repaired and checked by reading, not compiled, not run on hardware
//
#include <hip/hip_runtime.h>
#include <hip/hip_bf16.h>

// Problem constants (reference: B,C,H,W = 32,512,32,32)
#define BB   32
#define C_IN 512
#define C2   256
#define MM   1024   // H*W

typedef _Float16 half_t;
typedef __attribute__((ext_vector_type(16))) _Float16 v16h;
typedef __attribute__((ext_vector_type(8)))  _Float16 v8h;
typedef __attribute__((ext_vector_type(8)))  float    v8f;

#define WMMA_F16(a, b, c) \
    __builtin_amdgcn_wmma_f32_16x16x32_f16(false, (a), false, (b), (short)0, (c), false, false)

// ---- WMMA operand loaders (CDNA5 16-bit layouts, wave32) -------------------
// A (16x32, MxK): lane h=l>>4 holds K chunks {h*8..h*8+7} and {16+h*8..23+h*8}
__device__ __forceinline__ v16h load_a16(const half_t* base, int ld, int lane) {
    const int r = lane & 15, h = lane >> 4;
    const half_t* p = base + r * ld;
    v16h out;
    *((v8h*)&out)       = *(const v8h*)(p + h * 8);
    *(((v8h*)&out) + 1) = *(const v8h*)(p + 16 + h * 8);
    return out;
}
// B (32x16, KxN): lane = column (l&15), K chunks {h*8..h*8+7} and {16+h*8..23+h*8}
__device__ __forceinline__ v16h load_b16(const half_t* base, int ld, int lane) {
    const int c = lane & 15, h = lane >> 4;
    const half_t* p = base + c * ld;
    v16h out;
    *((v8h*)&out)       = *(const v8h*)(p + h * 8);
    *(((v8h*)&out) + 1) = *(const v8h*)(p + 16 + h * 8);
    return out;
}

// ---- Stage 0a: x [B][C][M] f32 -> xT [B][M][C] f16 (LDS tile transpose) ----
__global__ __launch_bounds__(256) void transpose_x_kernel(const float* __restrict__ x,
                                                          half_t* __restrict__ xT) {
    __shared__ float tile[32][33];
    const int bid = blockIdx.x;            // b*(16*32) + ct*32 + mt
    const int mt = bid & 31;
    const int ct = (bid >> 5) & 15;
    const int b  = bid >> 9;
    const int tx = threadIdx.x & 31, ty = threadIdx.x >> 5;   // ty in 0..7
    const float* xb = x + (size_t)b * C_IN * MM;
#pragma unroll
    for (int r = 0; r < 4; ++r) {
        const int c = ct * 32 + ty + r * 8;
        tile[ty + r * 8][tx] = xb[(size_t)c * MM + mt * 32 + tx];
    }
    __syncthreads();
    half_t* xTb = xT + (size_t)b * MM * C_IN;
#pragma unroll
    for (int r = 0; r < 4; ++r) {
        const int m = mt * 32 + ty + r * 8;
        xTb[(size_t)m * C_IN + ct * 32 + tx] = (half_t)tile[tx][ty + r * 8];
    }
}

// ---- Stage 0b: convert weights (all three are 256*512 = 131072 elems) ------
__global__ __launch_bounds__(256) void convert_w_kernel(const float* __restrict__ Wk,
                                                        const float* __restrict__ Wv,
                                                        const float* __restrict__ Wo,
                                                        half_t* __restrict__ Wkh,
                                                        half_t* __restrict__ Wvh,
                                                        half_t* __restrict__ Woh) {
    const int i = blockIdx.x * 256 + threadIdx.x;
    Wkh[i] = (half_t)Wk[i];
    Wvh[i] = (half_t)Wv[i];
    Woh[i] = (half_t)Wo[i];
}

// ---- Stage 1: k,v = W @ x  (wave tile: 64 m x 16 o, both K and V) ----------
// 8 WMMA per k-step vs 12 b128 loads (weights reused across 4 m-tiles).
// kT stored [B][M][c2] (m-major);  v stored in v_re^T layout [B][c2][M]:
//   v[o][m] has flat idx i=o*M+m; reference reshape makes v_re[n][c']=flat[n*c2+c']
//   -> vT[(i&255)*1024 + (i>>8)] so the attn@v GEMM B-operand is K-contiguous.
__global__ __launch_bounds__(128) void kv_gemm_kernel(const half_t* __restrict__ xT,
                                                      const half_t* __restrict__ Wkh,
                                                      const half_t* __restrict__ Wvh,
                                                      const float* __restrict__ bk,
                                                      const float* __restrict__ bv,
                                                      half_t* __restrict__ kT,
                                                      half_t* __restrict__ vT) {
    const int lane = threadIdx.x & 31;
    const int wave = blockIdx.x * 4 + (threadIdx.x >> 5);   // 8192 waves total
    const int b  = wave >> 8;            // 256 waves per batch
    const int r  = wave & 255;
    const int mg = r >> 4;               // 16 m-groups of 64 rows
    const int ot = r & 15;
    const int m0 = mg * 64;

    const half_t* A  = xT + ((size_t)b * MM + m0) * C_IN;
    const half_t* Bk = Wkh + (ot * 16) * C_IN;
    const half_t* Bv = Wvh + (ot * 16) * C_IN;

    v8f accK[4] = {};
    v8f accV[4] = {};
#pragma unroll
    for (int kk = 0; kk < 16; ++kk) {            // K = 512 = 16 * 32
        v16h a[4];
#pragma unroll
        for (int i = 0; i < 4; ++i) a[i] = load_a16(A + i * 16 * C_IN + kk * 32, C_IN, lane);
        const v16h b0 = load_b16(Bk + kk * 32, C_IN, lane);
        const v16h b1 = load_b16(Bv + kk * 32, C_IN, lane);
#pragma unroll
        for (int i = 0; i < 4; ++i) {
            accK[i] = WMMA_F16(a[i], b0, accK[i]);
            accV[i] = WMMA_F16(a[i], b1, accV[i]);
        }
    }
    const int h = lane >> 4, nl = lane & 15;
    const float bkv = bk[ot * 16 + nl];
    const float bvv = bv[ot * 16 + nl];
    half_t* kTb = kT + (size_t)b * MM * C2;
    half_t* vTb = vT + (size_t)b * C2 * MM;
#pragma unroll
    for (int i = 0; i < 4; ++i) {
#pragma unroll
        for (int g = 0; g < 8; ++g) {
            const int m = m0 + i * 16 + g + 8 * h;
            kTb[m * C2 + ot * 16 + nl] = (half_t)(accK[i][g] + bkv);
            const int o = ot * 16 + nl;
            const int ii = o * MM + m;                 // flat idx in v memory
            vTb[(ii & 255) * MM + (ii >> 8)] = (half_t)(accV[i][g] + bvv);
        }
    }
}

// ---- Stage 1b: s[b][m] = sum_c k[c][m] = row-sum of kT ----------------------
__global__ __launch_bounds__(256) void row_sum_kernel(const half_t* __restrict__ kT,
                                                      float* __restrict__ s) {
    const int t = blockIdx.x * 256 + threadIdx.x;       // 32768 rows total
    const v8h* r = (const v8h*)(kT + (size_t)t * C2);
    float acc = 0.f;
#pragma unroll
    for (int i = 0; i < 32; ++i) {
        const v8h c = r[i];
#pragma unroll
        for (int e = 0; e < 8; ++e) acc += (float)c[e];
    }
    s[t] = acc;
}

// ---- Stage 2: fused scores -> softmax -> attn @ v_re  ----------------------
// One wave per 16-row block: 16x1024 scores in LDS (f16, 32KB), shuffle
// reductions for row max/sum, then 16x256 output via 512 more WMMAs.
// s[] is staged into LDS via gfx1250 async global->LDS loads (ASYNCcnt).
__global__ __launch_bounds__(32) void attn_kernel(const half_t* __restrict__ kT,
                                                  const half_t* __restrict__ vT,
                                                  const float* __restrict__ s,
                                                  half_t* __restrict__ midT) {
    __shared__ half_t sc[16 * MM];    // 32 KB
    __shared__ float  sS[MM];         // 4 KB, filled asynchronously
    const int lane = threadIdx.x;
    const int wid  = blockIdx.x;                 // 32 batches * 64 blocks
    const int b  = wid >> 6;
    const int m0 = (wid & 63) * 16;
    const half_t* kTb = kT + (size_t)b * MM * C2;
    const half_t* vTb = vT + (size_t)b * C2 * MM;
    const float*  sb  = s + b * MM;
    const int h = lane >> 4, nl = lane & 15;

    // Async copy s[b][0..1023] (4KB) into LDS: 8 x b128 per lane.
#pragma unroll
    for (int t = 0; t < 8; ++t) {
        const int n4 = t * 128 + lane * 4;
        const unsigned long long ga = (unsigned long long)(const void*)(sb + n4);
        const unsigned int       la = (unsigned int)(unsigned long long)(const void*)(&sS[n4]);
        asm volatile("global_load_async_to_lds_b128 %0, %1, off"
                     :: "v"(la), "v"(ga) : "memory");
    }

    // A tiles: 16 rows of kT, all K=256 (held in 64 VGPRs, reused 64x)
    v16h a[8];
#pragma unroll
    for (int kk = 0; kk < 8; ++kk) a[kk] = load_a16(kTb + m0 * C2 + kk * 32, C2, lane);

    asm volatile("s_wait_asynccnt 0x0" ::: "memory");   // sS[] ready

    float smv[8], rmax[8];
#pragma unroll
    for (int g = 0; g < 8; ++g) { smv[g] = sS[m0 + g + 8 * h]; rmax[g] = -1e30f; }

    const float invM = 1.0f / (float)MM;
    const float invM2 = invM * invM;
    const float rscale = 0.0625f;                // c2^-0.5 = 1/16

    // Pass 1: scores S = ((k^T k)/M - s s^T/M^2) * scale  -> LDS (f16)
    for (int j = 0; j < 64; ++j) {
        __builtin_prefetch(kTb + ((j + 1) & 63) * 16 * C2, 0, 1);
        v8f acc = {};
#pragma unroll
        for (int kk = 0; kk < 8; ++kk) {
            const v16h bt = load_b16(kTb + (j * 16) * C2 + kk * 32, C2, lane);
            acc = WMMA_F16(a[kk], bt, acc);
        }
        const int n = j * 16 + nl;
        const float sn = sS[n];
#pragma unroll
        for (int g = 0; g < 8; ++g) {
            const float v = (acc[g] * invM - smv[g] * sn * invM2) * rscale;
            sc[(g + 8 * h) * MM + n] = (half_t)v;
            rmax[g] = fmaxf(rmax[g], v);
        }
    }
    // Row max across the 16 lanes that share a row group
#pragma unroll
    for (int g = 0; g < 8; ++g)
        for (int off = 1; off < 16; off <<= 1)
            rmax[g] = fmaxf(rmax[g], __shfl_xor(rmax[g], off, 16));

    // Pass 2: exponentiate in place, accumulate row sums
    float rsum[8];
#pragma unroll
    for (int g = 0; g < 8; ++g) rsum[g] = 0.f;
    for (int j = 0; j < 64; ++j) {
        const int n = j * 16 + nl;
#pragma unroll
        for (int g = 0; g < 8; ++g) {
            const int idx = (g + 8 * h) * MM + n;
            const float p = __expf((float)sc[idx] - rmax[g]);
            sc[idx] = (half_t)p;
            rsum[g] += p;
        }
    }
#pragma unroll
    for (int g = 0; g < 8; ++g) {
        for (int off = 1; off < 16; off <<= 1)
            rsum[g] += __shfl_xor(rsum[g], off, 16);
        rsum[g] = 1.0f / rsum[g];
    }

    // Pass 3: out = P @ v_re. kk outer so each LDS A-tile is loaded once and
    // reused by all 16 column tiles; 16 f32 accumulators (128 VGPRs) persist.
    v8f oacc[16];
#pragma unroll
    for (int ct = 0; ct < 16; ++ct) oacc[ct] = (v8f){};
    for (int kk = 0; kk < 32; ++kk) {            // K = 1024
        const v16h pa = load_a16(sc + kk * 32, MM, lane);
#pragma unroll
        for (int ct = 0; ct < 16; ++ct) {
            const v16h bt = load_b16(vTb + (ct * 16) * MM + kk * 32, MM, lane);
            oacc[ct] = WMMA_F16(pa, bt, oacc[ct]);
        }
    }
    half_t* midb = midT + (size_t)b * MM * C2;
#pragma unroll
    for (int ct = 0; ct < 16; ++ct) {
#pragma unroll
        for (int g = 0; g < 8; ++g) {
            const int m = m0 + g + 8 * h;
            const int c = ct * 16 + nl;
            const int i = m * C2 + c;            // flat idx in "out" memory
            // reference .view(B,c2,H,W) remap -> store B-operand layout for Wo GEMM
            midb[(i & 1023) * C2 + (i >> 10)] = (half_t)(oacc[ct][g] * rsum[g]);
        }
    }
}

// ---- Stage 3: out = Wo @ mid4 + bo + x  (wave tile: 32 oc x 64 m) ----------
__global__ __launch_bounds__(128) void out_gemm_kernel(const half_t* __restrict__ Woh,
                                                       const half_t* __restrict__ midT,
                                                       const float* __restrict__ bo,
                                                       const float* __restrict__ x,
                                                       float* __restrict__ out) {
    const int lane = threadIdx.x & 31;
    const int wave = blockIdx.x * 4 + (threadIdx.x >> 5);   // 8192 waves
    const int b   = wave >> 8;           // 256 waves per batch (16 oc x 16 m)
    const int r   = wave & 255;
    const int oc0 = (r >> 4) * 32;
    const int m0  = (r & 15) * 64;

    const half_t* A  = Woh + oc0 * C2;
    const half_t* Bb = midT + (size_t)b * MM * C2 + m0 * C2;
    v8f acc[2][4] = {};
#pragma unroll
    for (int kk = 0; kk < 8; ++kk) {             // K = 256
        v16h a[2], bt[4];
#pragma unroll
        for (int i = 0; i < 2; ++i) a[i]  = load_a16(A + i * 16 * C2 + kk * 32, C2, lane);
#pragma unroll
        for (int j = 0; j < 4; ++j) bt[j] = load_b16(Bb + j * 16 * C2 + kk * 32, C2, lane);
#pragma unroll
        for (int i = 0; i < 2; ++i)
#pragma unroll
            for (int j = 0; j < 4; ++j) acc[i][j] = WMMA_F16(a[i], bt[j], acc[i][j]);
    }
    const int h = lane >> 4, nl = lane & 15;
#pragma unroll
    for (int i = 0; i < 2; ++i) {
#pragma unroll
        for (int j = 0; j < 4; ++j) {
#pragma unroll
            for (int g = 0; g < 8; ++g) {
                const int oc = oc0 + i * 16 + g + 8 * h;
                const int m  = m0 + j * 16 + nl;
                const size_t o = (size_t)b * C_IN * MM + (size_t)oc * MM + m;
                out[o] = acc[i][j][g] + bo[oc] + x[o];
            }
        }
    }
}

// ---- Launch ----------------------------------------------------------------
extern "C" void kernel_launch(void* const* d_in, const int* in_sizes, int n_in,
                              void* d_out, int out_size, void* d_ws, size_t ws_size,
                              hipStream_t stream) {
    const float* x  = (const float*)d_in[0];
    const float* Wk = (const float*)d_in[1];
    const float* bk = (const float*)d_in[2];
    const float* Wv = (const float*)d_in[3];
    const float* bv = (const float*)d_in[4];
    const float* Wo = (const float*)d_in[5];
    const float* bo = (const float*)d_in[6];
    float* out = (float*)d_out;

    char* ws = (char*)d_ws;
    half_t* xT   = (half_t*)(ws);                       // 32 MB  [B][M][C] f16
    half_t* kT   = (half_t*)(ws + 33554432);            // 16 MB  [B][M][c2] f16
    half_t* vT   = (half_t*)(ws + 50331648);            // 16 MB  [B][c2][M] f16 (v_re^T)
    half_t* midT = (half_t*)(ws + 67108864);            // 16 MB  Wo-GEMM B layout
    float*  s    = (float*)(ws + 83886080);             // 128 KB [B][M]
    half_t* Wkh  = (half_t*)(ws + 84017152);            // 256 KB
    half_t* Wvh  = (half_t*)(ws + 84279296);            // 256 KB
    half_t* Woh  = (half_t*)(ws + 84541440);            // 256 KB  (~80.9 MB total)

    transpose_x_kernel<<<BB * 16 * 32, 256, 0, stream>>>(x, xT);
    convert_w_kernel<<<(C2 * C_IN) / 256, 256, 0, stream>>>(Wk, Wv, Wo, Wkh, Wvh, Woh);
    kv_gemm_kernel<<<2048, 128, 0, stream>>>(xT, Wkh, Wvh, bk, bv, kT, vT);
    row_sum_kernel<<<(BB * MM) / 256, 256, 0, stream>>>(kT, s);
    attn_kernel<<<BB * 64, 32, 0, stream>>>(kT, vT, s, midT);
    out_gemm_kernel<<<2048, 128, 0, stream>>>(Woh, midT, bo, x, out);
}
